// KANLayer_23184233464145
// MI455X (gfx1250) — matrix operand order repaired and checked by reading, not compiled, not run on hardware
//
#include <hip/hip_runtime.h>
#include <hip/hip_bf16.h>

// ---------------------------------------------------------------------------
// KAN layer for MI455X (gfx1250, wave32, WMMA + TDM).
//
// out[b,o] = sum_{j,k} basis(x[b,j])[k] * cp[o,j,k]
//   - uniform knots => cardinal cubic B-spline: 4 nonzero weights per x,
//     window m-3..m (m = floor(13x)), clipped to [0,10).
//   - contraction cast as f16 WMMA GEMM: M=batch, N=64 outputs,
//     K = 1024 j's x 16 (10 basis + 6 zero pad per j).
// HBM-bound (~37 MB => ~1.6us at 23.3 TB/s); basis is computed on the fly
// into LDS (never touches HBM), control points are converted to f16 once into
// d_ws and then streamed chunk-by-chunk into LDS by the Tensor Data Mover
// (with LDS padding done by TDM pad_enable to avoid bank conflicts).
// ---------------------------------------------------------------------------

typedef __attribute__((ext_vector_type(16))) _Float16 v16h;
typedef __attribute__((ext_vector_type(8)))  _Float16 v8h;
typedef __attribute__((ext_vector_type(8)))  float    v8f;
typedef __attribute__((ext_vector_type(4)))  unsigned int u32x4;
typedef __attribute__((ext_vector_type(8)))  int          i32x8;
typedef __attribute__((ext_vector_type(4)))  int          i32x4;

#if __has_builtin(__builtin_amdgcn_tensor_load_to_lds) && \
    __has_builtin(__builtin_amdgcn_s_wait_tensorcnt)
#define KAN_HAVE_TDM 1
#else
#define KAN_HAVE_TDM 0
#endif

namespace kan {
constexpr int BATCH  = 8192;
constexpr int J      = 1024;
constexpr int O      = 64;
constexpr int K13    = 13;       // stored control-point width
constexpr int NB     = 10;       // nonzero basis functions
constexpr int KPAD   = 16;       // per-j padded K for WMMA
constexpr int MT     = 128;      // rows per block (8 waves x 16)
constexpr int JC     = 8;        // j's per chunk (K-chunk = 128)
constexpr int KSPLIT = 4;        // split-K over j
constexpr int JSPL   = J / KSPLIT;      // 256
constexpr int NCHUNK = JSPL / JC;       // 32
constexpr int AROW   = JC * KPAD + 8;   // 136 halves: pad to dodge LDS bank conflicts
constexpr int BROW   = 24;              // halves per (j,o): 48B stride, 16B aligned
constexpr int CHUNK_ELEMS = JC * O * KPAD;          // 8192 f16 per chunk
constexpr int WAVE_ELEMS  = CHUNK_ELEMS / 8;        // 1024 f16 per wave TDM slice
constexpr int WAVE_LDS_B  = WAVE_ELEMS * 2 * BROW / KPAD;  // 3072 bytes (incl. pad)
}

__global__ __launch_bounds__(256) void kan_zero_out(float* __restrict__ out, int n) {
    int i = blockIdx.x * blockDim.x + threadIdx.x;
    if (i < n) out[i] = 0.0f;
}

// cp (o, j, 13) f32  ->  ws (j, o, 16) f16, zero padded k=13..15
__global__ __launch_bounds__(256) void kan_prep_b(const float* __restrict__ cp,
                                                  _Float16* __restrict__ ws) {
    int idx = blockIdx.x * blockDim.x + threadIdx.x;   // < J*O*KPAD = 1,048,576
    int j   = idx >> 10;          // O*KPAD = 1024
    int rem = idx & 1023;
    int o   = rem >> 4;
    int k   = rem & 15;
    float v = (k < kan::K13) ? cp[((size_t)o * kan::J + j) * kan::K13 + k] : 0.0f;
    ws[idx] = (_Float16)v;
}

template <bool USE_WS>
__global__ __launch_bounds__(256) void kan_main(const float* __restrict__ x,
                                                const float* __restrict__ cp,
                                                const _Float16* __restrict__ bws,
                                                float* __restrict__ out) {
    using namespace kan;
    __shared__ alignas(16) _Float16 Abuf[MT * AROW];       // ~34 KB
    __shared__ alignas(16) _Float16 Bbuf[JC * O * BROW];   // 24 KB

    const int t    = threadIdx.x;
    const int wv   = t >> 5;          // wave id 0..7 -> 16-row sub-tile
    const int lane = t & 31;
    const int l16  = lane & 15;
    const int hf   = lane >> 4;       // lane-half per WMMA fragment layout
    const int r0   = blockIdx.x * MT;
    const int jsb  = blockIdx.y * JSPL;

    v8f acc[4] = {};
    const int rA = wv * 16 + l16;

#if KAN_HAVE_TDM
    // Wave-uniform slice id for the per-wave TDM descriptor (SGPR-clean).
    const unsigned wvu = (unsigned)__builtin_amdgcn_readfirstlane(wv);
    // LDS byte offset of this wave's destination slice within Bbuf.
    const unsigned lds_slice =
        (unsigned)(__UINTPTR_TYPE__)(__attribute__((address_space(3))) void*)Bbuf
        + wvu * (unsigned)WAVE_LDS_B;
#endif

    for (int c = 0; c < NCHUNK; ++c) {
        const int jc = jsb + c * JC;
        bool b_done_by_tdm = false;

#if KAN_HAVE_TDM
        if (USE_WS) {
            // ---- B fill via Tensor Data Mover: ws chunk -> padded LDS -----
            // 1D tile of 1024 f16 per wave; TDM inserts 4 DWORDs of LDS pad
            // after every 8 DWORDs => BROW=24-half stride (bank-conflict-free
            // strided fragment reads later).
            unsigned long long ga = (unsigned long long)(__UINTPTR_TYPE__)bws
                + ((unsigned long long)jc * (O * KPAD) + wvu * WAVE_ELEMS) * 2ull;
            u32x4 g0;
            g0.x = 1u;                                   // count=1, no gather
            g0.y = lds_slice;                            // lds_addr (bytes)
            g0.z = (unsigned)(ga & 0xffffffffull);       // global_addr[31:0]
            g0.w = (unsigned)((ga >> 32) & 0x1ffffffull) // global_addr[56:32]
                   | (2u << 30);                         // type=2 ("image")
            i32x8 g1;
            g1[0] = (1 << 16)        // data_size = 2 bytes (f16)
                  | (1 << 20)        // pad_enable
                  | (2 << 22)        // pad_interval: every 8 DWORDs (32B)
                  | (3 << 25);       // pad_amount: 4 DWORDs (16B)
            g1[1] = (WAVE_ELEMS & 0xffff) << 16;         // tensor_dim0[15:0]
            g1[2] = ((WAVE_ELEMS >> 16) & 0xffff)        // tensor_dim0[31:16]
                  | (1 << 16);                           // tensor_dim1 = 1
            g1[3] = (WAVE_ELEMS & 0xffff) << 16;         // tile_dim0 = 1024
            g1[4] = 0;                                   // tile_dim1/2 unused
            g1[5] = WAVE_ELEMS;                          // tensor_dim0_stride
            g1[6] = 0;
            g1[7] = 0;
            i32x4 gz  = {0, 0, 0, 0};
            i32x8 gz8 = {0, 0, 0, 0, 0, 0, 0, 0};
            // 6-arg form (this toolchain): (g0, g1, g2, g3, g4, cpol)
            __builtin_amdgcn_tensor_load_to_lds(g0, g1, gz, gz, gz8, 0);
            b_done_by_tdm = true;
        }
#endif

        // ---- A fill: cardinal cubic B-spline weights -> f16 LDS tile ------
        #pragma unroll
        for (int p = 0; p < 4; ++p) {
            int idx = p * 256 + t;            // covers 128 rows x 8 j
            int row = idx >> 3;
            int jl  = idx & 7;
            float xv = x[(size_t)(r0 + row) * J + jc + jl];
            float s  = xv * 13.0f;
            int   m  = (int)s;                // x >= 0 => trunc == floor
            m = m > 12 ? 12 : m;
            float u  = s - (float)m;
            float v  = 1.0f - u;
            float u2 = u * u, u3 = u2 * u;
            float w0 = (v * v * v) * (1.0f / 6.0f);
            float w1 = (3.0f * u3 - 6.0f * u2 + 4.0f) * (1.0f / 6.0f);
            float w2 = (-3.0f * u3 + 3.0f * u2 + 3.0f * u + 1.0f) * (1.0f / 6.0f);
            float w3 = u3 * (1.0f / 6.0f);
            _Float16* ap = &Abuf[row * AROW + jl * KPAD];
            v8h z = {};
            *(v8h*)ap       = z;              // zero all 16 slots (2x ds_store_b128)
            *(v8h*)(ap + 8) = z;
            // scatter 4 weights into window [m-3, m] clipped to [0, NB)
            if (m - 3 >= 0)               ap[m - 3] = (_Float16)w0;
            if (m - 2 >= 0 && m - 2 < NB) ap[m - 2] = (_Float16)w1;
            if (m - 1 >= 0 && m - 1 < NB) ap[m - 1] = (_Float16)w2;
            if (m < NB)                   ap[m]     = (_Float16)w3;
        }

        // ---- B fill fallback paths ---------------------------------------
        if (!b_done_by_tdm) {
            if (USE_WS) {
                #pragma unroll
                for (int p = 0; p < 4; ++p) {
                    int u_  = p * 256 + t;        // 1024 x 16B units
                    int jl  = u_ >> 7;
                    int rem = u_ & 127;
                    int o   = rem >> 1;
                    int h   = rem & 1;
                    v8h vb = *(const v8h*)&bws[((size_t)(jc + jl) * O + o) * KPAD + h * 8];
                    *(v8h*)&Bbuf[(jl * O + o) * BROW + h * 8] = vb;
                }
            } else {
                #pragma unroll 4
                for (int p = 0; p < 32; ++p) {
                    int idx = p * 256 + t;        // JC*O*16 = 8192 halves
                    int jl  = idx >> 10;
                    int rem = idx & 1023;
                    int o   = rem >> 4;
                    int k   = rem & 15;
                    float v = (k < K13) ? cp[((size_t)o * J + jc + jl) * K13 + k] : 0.0f;
                    Bbuf[(jl * O + o) * BROW + k] = (_Float16)v;
                }
            }
        }

        // prefetch next chunk of x into caches (global_prefetch_b8)
        if (c + 1 < NCHUNK) {
            __builtin_prefetch(&x[(size_t)(r0 + (t >> 3)) * J + jc + JC + (t & 7)], 0, 1);
        }

#if KAN_HAVE_TDM
        if (b_done_by_tdm) __builtin_amdgcn_s_wait_tensorcnt(0);
#endif
        __syncthreads();

        // ---- compute: 4 K-steps (2 j's each) x 4 N-tiles ------------------
        #pragma unroll
        for (int ks = 0; ks < 4; ++ks) {
            const int j0 = 2 * ks, j1 = j0 + 1;
            // A 16x32 f16 fragment: lanes 0-15 carry K-groups {0-7,16-23},
            // lanes 16-31 carry {8-15,24-31}; K 0..15 = j0, 16..31 = j1.
            v8h al = *(const v8h*)&Abuf[rA * AROW + j0 * KPAD + hf * 8];
            v8h ah = *(const v8h*)&Abuf[rA * AROW + j1 * KPAD + hf * 8];
            // B 32x16 f16 fragment: same K-groups as A (lanes 0-15 {0-7,16-23},
            // lanes 16-31 {8-15,24-31}); N = lane%16.
            // Batch all 8 b128 loads so one dscnt wait covers 4 WMMAs.
            const int jh = hf ? j1 : j0;
            v8h bl[4], bh[4];
            #pragma unroll
            for (int nt = 0; nt < 4; ++nt) {
                const _Float16* bp = &Bbuf[(j0 * O + nt * 16 + l16) * BROW + hf * 8];
                bl[nt] = *(const v8h*)bp;
                bh[nt] = *(const v8h*)(bp + O * BROW);
            }
            v16h a;
            #pragma unroll
            for (int i = 0; i < 8; ++i) { a[i] = al[i]; a[i + 8] = ah[i]; }
            #pragma unroll
            for (int nt = 0; nt < 4; ++nt) {
                v16h b;
                #pragma unroll
                for (int i = 0; i < 8; ++i) { b[i] = bl[nt][i]; b[i + 8] = bh[nt][i]; }
                acc[nt] = __builtin_amdgcn_wmma_f32_16x16x32_f16(
                    false, a, false, b, (short)0, acc[nt], false, false);
            }
        }

        __syncthreads();
    }

    // ---- epilogue: C/D layout lanes 0-15 -> M=i, lanes 16-31 -> M=8+i -----
    #pragma unroll
    for (int nt = 0; nt < 4; ++nt) {
        #pragma unroll
        for (int i = 0; i < 8; ++i) {
            int row = r0 + wv * 16 + hf * 8 + i;
            int col = nt * 16 + l16;
            unsafeAtomicAdd(&out[(size_t)row * kan::O + col], acc[nt][i]);
        }
    }
}

extern "C" void kernel_launch(void* const* d_in, const int* in_sizes, int n_in,
                              void* d_out, int out_size, void* d_ws, size_t ws_size,
                              hipStream_t stream) {
    using namespace kan;
    const float* x  = (const float*)d_in[0];
    // d_in[1] = knots: linspace(0,1,14) => uniform, spacing 1/13 hardcoded.
    const float* cp = (const float*)d_in[2];
    float* out = (float*)d_out;

    // zero output (split-K accumulates with f32 atomics)
    kan_zero_out<<<(BATCH * O + 255) / 256, 256, 0, stream>>>(out, BATCH * O);

    const size_t ws_needed = (size_t)J * O * KPAD * 2;  // 2 MB of f16
    dim3 grid(BATCH / MT, KSPLIT);
    if (ws_size >= ws_needed) {
        kan_prep_b<<<(J * O * KPAD) / 256, 256, 0, stream>>>(cp, (_Float16*)d_ws);
        kan_main<true><<<grid, 256, 0, stream>>>(x, cp, (const _Float16*)d_ws, out);
    } else {
        kan_main<false><<<grid, 256, 0, stream>>>(x, cp, nullptr, out);
    }
}
